// Net_76982993813580
// MI455X (gfx1250) — matrix operand order repaired and checked by reading, not compiled, not run on hardware
//
#include <hip/hip_runtime.h>
#include <hip/hip_bf16.h>
#include <cmath>

// ---------------- model dims ----------------
#define NN    98304       // nodes = 4096*24
#define NE    1572864     // edges
#define FIN   5
#define HIDN  32
#define LH    118         // LSTM hidden
#define CC    236         // 2*LH
#define TT    24
#define BB    4096
#define GPAD  480         // 4*118=472 -> 480
#define HPAD  128         // 118 -> 128
#define K1PAD 256         // 236 -> 256
#define FCK   2848        // 2832 -> 2848
#define ALPHA_ 0.1f

typedef __attribute__((ext_vector_type(16))) __bf16 v16bf;
typedef __attribute__((ext_vector_type(8)))  __bf16 v8bf;
typedef __attribute__((ext_vector_type(8)))  float  v8f;

__device__ __forceinline__ float sigm(float x) { return 1.0f / (1.0f + __expf(-x)); }

__device__ __forceinline__ v16bf ldfragA(const __bf16* p) {
    v8bf lo = *(const v8bf*)(p);
    v8bf hi = *(const v8bf*)(p + 16);      // second contiguous K-run of this lane
    return __builtin_shufflevector(lo, hi, 0,1,2,3,4,5,6,7,8,9,10,11,12,13,14,15);
}
__device__ __forceinline__ v16bf ldfragB(const __bf16* p) {
    v8bf lo = *(const v8bf*)(p);
    v8bf hi = *(const v8bf*)(p + 16);      // second contiguous K-run of this lane
    return __builtin_shufflevector(lo, hi, 0,1,2,3,4,5,6,7,8,9,10,11,12,13,14,15);
}

// ================= generic WMMA GEMM (2x2 register-blocked) =================
// A: [M x K] row-major bf16 (K mult 32, M mult 32)
// B: [N x K] column-major-of-logical-[KxN] bf16 (B[n*K + k]), N mult 32
// out: [M x Nout] f32 ; res (optional): [M x Nout] f32 added before relu
__global__ void wmma_gemm(const __bf16* __restrict__ A, const __bf16* __restrict__ B,
                          const float* __restrict__ bias, const float* __restrict__ res,
                          float* __restrict__ out, int M, int N, int K, int Nout, int relu)
{
    const int lane  = threadIdx.x & 31;
    const int wave  = blockIdx.x * (blockDim.x >> 5) + (threadIdx.x >> 5);
    const int nwav  = gridDim.x * (blockDim.x >> 5);
    const int stN   = N >> 5;                 // super-tiles (32 wide)
    const int stM   = M >> 5;
    const int tiles = stM * stN;
    const int mloc  = lane & 15;
    const int half  = lane >> 4;

    for (int tid = wave; tid < tiles; tid += nwav) {
        const int mt0 = (tid / stN) << 1;     // 16-row tile indices
        const int nt0 = (tid - (tid / stN) * stN) << 1;
        const __bf16* a0 = A + (size_t)((mt0 << 4) + mloc) * K + (half << 3);
        const __bf16* a1 = a0 + (size_t)16 * K;
        const __bf16* b0 = B + (size_t)((nt0 << 4) + mloc) * K + (half << 3);
        const __bf16* b1 = b0 + (size_t)16 * K;
        v8f c00 = {}, c01 = {}, c10 = {}, c11 = {};
        for (int k0 = 0; k0 < K; k0 += 32) {
            v16bf fa0 = ldfragA(a0 + k0);
            v16bf fa1 = ldfragA(a1 + k0);
            v16bf fb0 = ldfragB(b0 + k0);
            v16bf fb1 = ldfragB(b1 + k0);
            if (k0 + 32 < K) {                            // global_prefetch next K block
                __builtin_prefetch(a0 + k0 + 32, 0, 0);
                __builtin_prefetch(a1 + k0 + 32, 0, 0);
            }
            c00 = __builtin_amdgcn_wmma_f32_16x16x32_bf16(false, fa0, false, fb0, (short)0, c00, false, false);
            c01 = __builtin_amdgcn_wmma_f32_16x16x32_bf16(false, fa0, false, fb1, (short)0, c01, false, false);
            c10 = __builtin_amdgcn_wmma_f32_16x16x32_bf16(false, fa1, false, fb0, (short)0, c10, false, false);
            c11 = __builtin_amdgcn_wmma_f32_16x16x32_bf16(false, fa1, false, fb1, (short)0, c11, false, false);
        }
        #pragma unroll
        for (int ni = 0; ni < 2; ++ni) {
            const int n = ((nt0 + ni) << 4) + mloc;
            if (n >= Nout) continue;
            const float bv = bias ? bias[n] : 0.0f;
            #pragma unroll
            for (int mi = 0; mi < 2; ++mi) {
                const v8f& acc = ni ? (mi ? c11 : c01) : (mi ? c10 : c00);
                #pragma unroll
                for (int r = 0; r < 8; ++r) {
                    const int m = ((mt0 + mi) << 4) + (half << 3) + r;
                    float v = acc[r] + bv;
                    if (res)  v += res[(size_t)m * Nout + n];
                    if (relu) v = fmaxf(v, 0.0f);
                    out[(size_t)m * Nout + n] = v;
                }
            }
        }
    }
}

// ================= small utility kernels =================
__global__ void fill_f32(float* p, float v, int n) {
    int i = blockIdx.x * blockDim.x + threadIdx.x;
    if (i < n) p[i] = v;
}
__global__ void fill_bf16(__bf16* p, int n) {
    int i = blockIdx.x * blockDim.x + threadIdx.x;
    if (i < n) p[i] = (__bf16)0.0f;
}
// deg[col[e]] += w[e]
__global__ void deg_scatter(const int* __restrict__ ei, const float* __restrict__ ea,
                            float* __restrict__ deg, int ne) {
    int e = blockIdx.x * blockDim.x + threadIdx.x;
    if (e < ne) atomicAdd(&deg[ei[ne + e]], ea[e]);
}
__global__ void dinv_k(const float* __restrict__ deg, float* __restrict__ dinv, int n) {
    int i = blockIdx.x * blockDim.x + threadIdx.x;
    if (i < n) { float d = deg[i]; dinv[i] = d > 0.0f ? rsqrtf(fmaxf(d, 1e-12f)) : 0.0f; }
}
// x0 = data@W(5x32)+b ; x = relu(x0)
__global__ void lin0_k(const float* __restrict__ data, const float* __restrict__ w,
                       const float* __restrict__ b, float* __restrict__ x0,
                       float* __restrict__ x) {
    int id = blockIdx.x * blockDim.x + threadIdx.x;
    if (id >= NN * HIDN) return;
    int i = id >> 5, c = id & 31;
    float acc = b[c];
    #pragma unroll
    for (int k = 0; k < FIN; ++k) acc += data[i * FIN + k] * w[k * HIDN + c];
    x0[id] = acc; x[id] = fmaxf(acc, 0.0f);
}
// h = self-loop contribution  (also zero-initializes h)
__global__ void agg_self(const float* __restrict__ x, const float* __restrict__ dinv,
                         float* __restrict__ h) {
    int id = blockIdx.x * blockDim.x + threadIdx.x;
    if (id >= NN * HIDN) return;
    int i = id >> 5;
    float d = dinv[i];
    h[id] = d * d * x[id];
}
// h[col] += dinv[row]*w*dinv[col] * x[row]   (4 threads/edge, 8 channels each)
__global__ void agg_edges(const int* __restrict__ ei, const float* __restrict__ ea,
                          const float* __restrict__ dinv, const float* __restrict__ x,
                          float* __restrict__ h) {
    int t = blockIdx.x * blockDim.x + threadIdx.x;
    if (t >= NE * 4) return;
    int e = t >> 2, c0 = (t & 3) * 8;
    int r = ei[e], c = ei[NE + e];
    float wn = dinv[r] * ea[e] * dinv[c];
    const float* xs = x + (size_t)r * HIDN + c0;
    float* hd = h + (size_t)c * HIDN + c0;
    #pragma unroll
    for (int k = 0; k < 8; ++k) atomicAdd(&hd[k], wn * xs[k]);
}
__global__ void combine_k(float* __restrict__ h, const float* __restrict__ x0, int n) {
    int i = blockIdx.x * blockDim.x + threadIdx.x;
    if (i < n) h[i] = (1.0f - ALPHA_) * h[i] + ALPHA_ * x0[i];
}
// f32 [rows x cin] -> bf16 [rows x kpad] zero-padded columns
__global__ void to_bf16_pad(const float* __restrict__ in, __bf16* __restrict__ out,
                            int rows, int cin, int kpad) {
    int id = blockIdx.x * blockDim.x + threadIdx.x;
    if (id >= rows * kpad) return;
    int r = id / kpad, k = id - r * kpad;
    out[id] = (k < cin) ? (__bf16)in[(size_t)r * cin + k] : (__bf16)0.0f;
}
// W' = (1-beta)*I + beta*W ; stored col-major [32][32]
__global__ void make_wprime(const float* __restrict__ w, __bf16* __restrict__ wp, float beta) {
    int id = blockIdx.x * blockDim.x + threadIdx.x;
    if (id >= HIDN * HIDN) return;
    int n = id >> 5, k = id & 31;
    float v = beta * w[k * HIDN + n] + ((k == n) ? (1.0f - beta) : 0.0f);
    wp[n * HIDN + k] = (__bf16)v;
}
// pack weight W stored [N0 x K0] (torch [out,in], used as x@W^T) -> B[n*Kpad+k]
__global__ void pack_wT(const float* __restrict__ w, __bf16* __restrict__ B,
                        int N0, int K0, int Npad, int Kpad) {
    int id = blockIdx.x * blockDim.x + threadIdx.x;
    if (id >= Npad * Kpad) return;
    int n = id / Kpad, k = id - n * Kpad;
    B[id] = (n < N0 && k < K0) ? (__bf16)w[(size_t)n * K0 + k] : (__bf16)0.0f;
}
// pack weight W stored [K0 x N0] (used as x@W) -> B[n*Kpad+k]
__global__ void pack_w(const float* __restrict__ w, __bf16* __restrict__ B,
                       int K0, int N0, int Npad, int Kpad) {
    int id = blockIdx.x * blockDim.x + threadIdx.x;
    if (id >= Npad * Kpad) return;
    int n = id / Kpad, k = id - n * Kpad;
    B[id] = (n < N0 && k < K0) ? (__bf16)w[(size_t)k * N0 + n] : (__bf16)0.0f;
}
__global__ void bias_sum(const float* __restrict__ b1, const float* __restrict__ b2,
                         float* __restrict__ o, int n, int npad) {
    int i = blockIdx.x * blockDim.x + threadIdx.x;
    if (i < npad) o[i] = (i < n) ? (b1[i] + b2[i]) : 0.0f;
}
// node order [b*24+t][32] f32 -> seq order [t*4096+b][32] bf16
__global__ void permute_seq(const float* __restrict__ x, __bf16* __restrict__ o) {
    int id = blockIdx.x * blockDim.x + threadIdx.x;
    if (id >= NN * HIDN) return;
    int r = id >> 5, c = id & 31;
    int t = r / BB, b = r - t * BB;
    o[id] = (__bf16)x[(size_t)(b * TT + t) * HIDN + c];
}
// LSTM elementwise cell. g:[4096 x GPAD] gate order i,f,g,o (118 each)
__global__ void lstm_cell_k(const float* __restrict__ g, float* __restrict__ c,
                            __bf16* __restrict__ hpad, float* __restrict__ seqout,
                            int t, int doff) {
    int id = blockIdx.x * blockDim.x + threadIdx.x;
    if (id >= BB * LH) return;
    int b = id / LH, j = id - b * LH;
    const float* gr = g + (size_t)b * GPAD;
    float gi = gr[j], gf = gr[LH + j], gg = gr[2 * LH + j], go = gr[3 * LH + j];
    float cn = sigm(gf) * c[id] + sigm(gi) * tanhf(gg);
    float hn = sigm(go) * tanhf(cn);
    c[id] = cn;
    hpad[(size_t)b * HPAD + j] = (__bf16)hn;
    seqout[(size_t)(t * BB + b) * CC + doff + j] = hn;
}
// x[b,c,w] view of seqout (raw reshape [B,T,236] -> [B,236,24])
__device__ __forceinline__ float xval(const float* s, int b, int c, int w) {
    int idx = c * TT + w;
    int t = idx / CC, f = idx - t * CC;
    return s[(size_t)(t * BB + b) * CC + f];
}
__global__ void ca_pool(const float* __restrict__ s, float* __restrict__ avg,
                        float* __restrict__ mx) {
    int id = blockIdx.x * blockDim.x + threadIdx.x;
    if (id >= BB * CC) return;
    int b = id / CC, c = id - b * CC;
    float a = 0.0f, m = -3.4e38f;
    for (int w = 0; w < TT; ++w) { float v = xval(s, b, c, w); a += v; m = fmaxf(m, v); }
    avg[id] = a * (1.0f / TT); mx[id] = m;
}
__global__ void ca_mlp1(const float* __restrict__ v, const float* __restrict__ w1,
                        float* __restrict__ hid) {          // w1: [14 x 236]
    int id = blockIdx.x * blockDim.x + threadIdx.x;
    if (id >= BB * 14) return;
    int b = id / 14, hh = id - b * 14;
    float a = 0.0f;
    for (int c = 0; c < CC; ++c) a += v[(size_t)b * CC + c] * w1[hh * CC + c];
    hid[id] = fmaxf(a, 0.0f);
}
__global__ void ca_mlp2(const float* __restrict__ ha, const float* __restrict__ hm,
                        const float* __restrict__ w2, float* __restrict__ ca) { // w2: [236 x 14]
    int id = blockIdx.x * blockDim.x + threadIdx.x;
    if (id >= BB * CC) return;
    int b = id / CC, c = id - b * CC;
    float a = 0.0f;
    #pragma unroll
    for (int hh = 0; hh < 14; ++hh)
        a += (ha[(size_t)b * 14 + hh] + hm[(size_t)b * 14 + hh]) * w2[c * 14 + hh];
    ca[id] = sigm(a);
}
__global__ void sa_stats(const float* __restrict__ s, const float* __restrict__ ca,
                         float* __restrict__ smean, float* __restrict__ smax) {
    int id = blockIdx.x * blockDim.x + threadIdx.x;
    if (id >= BB * TT) return;
    int b = id / TT, w = id - b * TT;
    float a = 0.0f, m = -3.4e38f;
    for (int c = 0; c < CC; ++c) {
        float v = ca[(size_t)b * CC + c] * xval(s, b, c, w);
        a += v; m = fmaxf(m, v);
    }
    smean[id] = a * (1.0f / CC); smax[id] = m;
}
// 7-tap conv over h (width dim has only kw=3 in-bounds), saw: [1][2][7][7]
__global__ void sa_conv(const float* __restrict__ smean, const float* __restrict__ smax,
                        const float* __restrict__ saw, const float* __restrict__ sab,
                        float* __restrict__ so) {
    int id = blockIdx.x * blockDim.x + threadIdx.x;
    if (id >= BB * TT) return;
    int b = id / TT, hpos = id - b * TT;
    float a = sab[0];
    #pragma unroll
    for (int kh = 0; kh < 7; ++kh) {
        int hh = hpos + kh - 3;
        if (hh >= 0 && hh < TT) {
            a += smean[(size_t)b * TT + hh] * saw[kh * 7 + 3];
            a += smax [(size_t)b * TT + hh] * saw[49 + kh * 7 + 3];
        }
    }
    so[id] = a;
}
// apply ca & sigmoid(sa), maxpool (2,1), flatten c*12+p -> bf16 [4096 x FCK]
__global__ void pool_pack(const float* __restrict__ s, const float* __restrict__ ca,
                          const float* __restrict__ so, __bf16* __restrict__ out) {
    int id = blockIdx.x * blockDim.x + threadIdx.x;
    if (id >= BB * FCK) return;
    int b = id / FCK, col = id - b * FCK;
    if (col >= CC * (TT / 2)) { out[id] = (__bf16)0.0f; return; }
    int c = col / (TT / 2), p = col - c * (TT / 2);
    float cav = ca[(size_t)b * CC + c];
    float v0 = sigm(so[(size_t)b * TT + 2 * p])     * cav * xval(s, b, c, 2 * p);
    float v1 = sigm(so[(size_t)b * TT + 2 * p + 1]) * cav * xval(s, b, c, 2 * p + 1);
    out[id] = (__bf16)fmaxf(v0, v1);
}

// ================= host side =================
static inline size_t al256(size_t x) { return (x + 255) & ~(size_t)255; }
#define L1D(kern, n, ...) kern<<<((n) + 255) / 256, 256, 0, stream>>>(__VA_ARGS__)

extern "C" void kernel_launch(void* const* d_in, const int* in_sizes, int n_in,
                              void* d_out, int out_size, void* d_ws, size_t ws_size,
                              hipStream_t stream) {
    const float* data   = (const float*)d_in[0];
    const int*   ei     = (const int*)  d_in[1];
    const float* ea     = (const float*)d_in[2];
    const float* lin0_w = (const float*)d_in[3];
    const float* lin0_b = (const float*)d_in[4];
    const float* gcn_w[4] = { (const float*)d_in[5], (const float*)d_in[6],
                              (const float*)d_in[7], (const float*)d_in[8] };
    // LSTM params at 9 + (l*2+d)*4 + {0:Wih,1:Whh,2:bih,3:bhh}
    const float* ca_w1  = (const float*)d_in[25];
    const float* ca_w2  = (const float*)d_in[26];
    const float* sa_w   = (const float*)d_in[27];
    const float* sa_b   = (const float*)d_in[28];
    const float* lin1_w = (const float*)d_in[29];
    const float* lin1_b = (const float*)d_in[30];
    const float* lin2_w = (const float*)d_in[31];
    const float* lin2_b = (const float*)d_in[32];
    float* outp = (float*)d_out;

    char* W = (char*)d_ws; size_t off = 0;
    auto alloc = [&](size_t bytes) { void* p = W + off; off = al256(off + bytes); return p; };
    float*  deg    = (float*) alloc((size_t)NN * 4);
    float*  dinv   = (float*) alloc((size_t)NN * 4);
    float*  x0     = (float*) alloc((size_t)NN * HIDN * 4);
    float*  x      = (float*) alloc((size_t)NN * HIDN * 4);
    float*  h      = (float*) alloc((size_t)NN * HIDN * 4);
    __bf16* hbf    = (__bf16*)alloc((size_t)NN * HIDN * 2);
    __bf16* wp     = (__bf16*)alloc((size_t)HIDN * HIDN * 2);
    __bf16* wih[4]; __bf16* whh[4]; float* bsum[4];
    for (int i = 0; i < 4; ++i) {
        wih[i]  = (__bf16*)alloc((size_t)GPAD * K1PAD * 2);
        whh[i]  = (__bf16*)alloc((size_t)GPAD * HPAD * 2);
        bsum[i] = (float*) alloc((size_t)GPAD * 4);
    }
    __bf16* seqbf  = (__bf16*)alloc((size_t)NN * K1PAD * 2);
    float*  gates  = (float*) alloc((size_t)NN * GPAD * 4);
    float*  gbuf   = (float*) alloc((size_t)BB * GPAD * 4);
    __bf16* hpad   = (__bf16*)alloc((size_t)BB * HPAD * 2);
    float*  cst    = (float*) alloc((size_t)BB * LH * 4);
    float*  seqout = (float*) alloc((size_t)NN * CC * 4);
    float*  avg    = (float*) alloc((size_t)BB * CC * 4);
    float*  mx     = (float*) alloc((size_t)BB * CC * 4);
    float*  hav    = (float*) alloc((size_t)BB * 14 * 4);
    float*  hmx    = (float*) alloc((size_t)BB * 14 * 4);
    float*  cab    = (float*) alloc((size_t)BB * CC * 4);
    float*  smean  = (float*) alloc((size_t)BB * TT * 4);
    float*  smax   = (float*) alloc((size_t)BB * TT * 4);
    float*  sconv  = (float*) alloc((size_t)BB * TT * 4);
    __bf16* poolbf = (__bf16*)alloc((size_t)BB * FCK * 2);
    __bf16* l1wbf  = (__bf16*)alloc((size_t)32 * FCK * 2);
    float*  fc1    = (float*) alloc((size_t)BB * 24 * 4);
    __bf16* fc1bf  = (__bf16*)alloc((size_t)BB * 32 * 2);
    __bf16* l2wbf  = (__bf16*)alloc((size_t)32 * 32 * 2);
    (void)ws_size; (void)n_in; (void)in_sizes; (void)out_size;

    auto gemm = [&](const __bf16* A, const __bf16* B, const float* bias, const float* res,
                    float* out, int M, int N, int K, int Nout, int relu) {
        int tiles = (M >> 5) * (N >> 5);        // 32x32 super-tiles per wave
        int blocks = (tiles + 7) / 8;           // 8 waves (wave32) per 256-thread block
        if (blocks < 1) blocks = 1;
        wmma_gemm<<<blocks, 256, 0, stream>>>(A, B, bias, res, out, M, N, K, Nout, relu);
    };

    // ---------- GCN norm ----------
    L1D(fill_f32, NN, deg, 1.0f, NN);                  // self-loop weight
    L1D(deg_scatter, NE, ei, ea, deg, NE);
    L1D(dinv_k, NN, deg, dinv, NN);
    L1D(lin0_k, NN * HIDN, data, lin0_w, lin0_b, x0, x);

    // ---------- 4 GCNII layers ----------
    const int layers[4] = {1, 2, 4, 8};
    for (int i = 0; i < 4; ++i) {
        float beta = logf(0.5f / (float)layers[i] + 1.0f);
        L1D(agg_self, NN * HIDN, x, dinv, h);
        L1D(agg_edges, NE * 4, ei, ea, dinv, x, h);
        L1D(combine_k, NN * HIDN, h, x0, NN * HIDN);
        L1D(to_bf16_pad, NN * HIDN, h, hbf, NN, HIDN, HIDN);
        L1D(make_wprime, HIDN * HIDN, gcn_w[i], wp, beta);
        gemm(hbf, wp, nullptr, nullptr, x, NN, HIDN, HIDN, HIDN, 1);
    }

    // ---------- pack LSTM weights ----------
    for (int l = 0; l < 2; ++l)
        for (int d = 0; d < 2; ++d) {
            int idx = l * 2 + d, base = 9 + idx * 4;
            int kin = (l == 0) ? HIDN : CC, kpad = (l == 0) ? HIDN : K1PAD;
            L1D(pack_wT, GPAD * kpad, (const float*)d_in[base + 0], wih[idx], 4 * LH, kin, GPAD, kpad);
            L1D(pack_wT, GPAD * HPAD, (const float*)d_in[base + 1], whh[idx], 4 * LH, LH, GPAD, HPAD);
            L1D(bias_sum, GPAD, (const float*)d_in[base + 2], (const float*)d_in[base + 3],
                bsum[idx], 4 * LH, GPAD);
        }

    // ---------- biLSTM, 2 layers ----------
    L1D(permute_seq, NN * HIDN, x, seqbf);             // layer0 input, K=32
    for (int l = 0; l < 2; ++l) {
        int K = (l == 0) ? HIDN : K1PAD;
        if (l == 1) L1D(to_bf16_pad, NN * K1PAD, seqout, seqbf, NN, CC, K1PAD);
        for (int d = 0; d < 2; ++d) {
            int idx = l * 2 + d;
            gemm(seqbf, wih[idx], bsum[idx], nullptr, gates, NN, GPAD, K, GPAD, 0);
            L1D(fill_bf16, BB * HPAD, hpad, BB * HPAD);
            L1D(fill_f32, BB * LH, cst, 0.0f, BB * LH);
            for (int s = 0; s < TT; ++s) {
                int t = d ? (TT - 1 - s) : s;
                gemm(hpad, whh[idx], nullptr, gates + (size_t)t * BB * GPAD,
                     gbuf, BB, GPAD, HPAD, GPAD, 0);
                L1D(lstm_cell_k, BB * LH, gbuf, cst, hpad, seqout, t, d * LH);
            }
        }
    }

    // ---------- CBAM ----------
    L1D(ca_pool, BB * CC, seqout, avg, mx);
    L1D(ca_mlp1, BB * 14, avg, ca_w1, hav);
    L1D(ca_mlp1, BB * 14, mx, ca_w1, hmx);
    L1D(ca_mlp2, BB * CC, hav, hmx, ca_w2, cab);
    L1D(sa_stats, BB * TT, seqout, cab, smean, smax);
    L1D(sa_conv, BB * TT, smean, smax, sa_w, sa_b, sconv);
    L1D(pool_pack, BB * FCK, seqout, cab, sconv, poolbf);

    // ---------- heads ----------
    L1D(pack_w, 32 * FCK, lin1_w, l1wbf, CC * (TT / 2), 24, 32, FCK);
    gemm(poolbf, l1wbf, lin1_b, nullptr, fc1, BB, 32, FCK, 24, 1);
    L1D(to_bf16_pad, BB * 32, fc1, fc1bf, BB, 24, 32);
    L1D(pack_w, 32 * 32, lin2_w, l2wbf, 24, 24, 32, 32);
    gemm(fc1bf, l2wbf, lin2_b, nullptr, outp, BB, 32, 32, 24, 1);
}
